// Attention_20066087207334
// MI455X (gfx1250) — matrix operand order repaired and checked by reading, not compiled, not run on hardware
//
#include <hip/hip_runtime.h>
#include <hip/hip_bf16.h>

// ---------------------------------------------------------------------------
// MI455X (gfx1250) attention: all matmuls on v_wmma_f32_16x16x32_f16 (wave32).
//   K1: Q/K/V = X @ W^T  (f32 in, f16 out to workspace), LDS-tiled WMMA GEMM.
//   K2: flash-style causal attention; 4 waves/block share one (b,h) and one
//       async-staged V chunk (GLOBAL_LOAD_ASYNC_TO_LDS_B128 + s_wait_asynccnt),
//       each wave owning a 16-row q-tile.
// Fragment layouts used here (contiguous k; A and B share it):
//   A 16x32 f16 : lane<16 -> M=lane,   halves 0..15 = K 0..15
//                 lane>=16-> M=lane-16, halves 0..15 = K 16..31
//   B 32x16 f16 : lane<16 -> N=lane,   halves 0..15 = K 0..15
//                 lane>=16-> N=lane-16, halves 0..15 = K 16..31
//   C/D 16x16 f32: VGPR g, lane l -> (M = g + 8*(l>>4), N = l&15)
// L2 note: f16 Q/K/V total 48 MB < 192 MB global L2, so the per-q-tile K/V
// re-reads are L2 hits; HBM traffic stays ~first-touch (~200 MB total).
// ---------------------------------------------------------------------------

typedef __attribute__((ext_vector_type(16))) _Float16 v16h;
typedef __attribute__((ext_vector_type(8)))  _Float16 v8h;
typedef __attribute__((ext_vector_type(8)))  float    v8f;
typedef __attribute__((ext_vector_type(4)))  float    v4f;

constexpr int kB   = 4;
constexpr int kT   = 2048;
constexpr int kE   = 1024;
constexpr int kH   = 16;
constexpr int kD   = 64;        // head dim

// ---- Kernel 1: projection GEMM  Y(BT x E) = X(BT x E) @ W^T, stored f16 ----
constexpr int BM = 64, BN = 64, BK = 32;

__global__ __launch_bounds__(128)
void proj_kernel(const float* __restrict__ Xq, const float* __restrict__ Xkv,
                 const float* __restrict__ Wq, const float* __restrict__ Wk,
                 const float* __restrict__ Wv,
                 _Float16* __restrict__ Qh, _Float16* __restrict__ Kh,
                 _Float16* __restrict__ Vh)
{
    const int which = blockIdx.z;                 // 0:Q 1:K 2:V
    const float* X  = (which == 0) ? Xq : Xkv;
    const float* W  = (which == 0) ? Wq : (which == 1 ? Wk : Wv);
    _Float16* Dst   = (which == 0) ? Qh : (which == 1 ? Kh : Vh);

    const int m0 = blockIdx.x * BM;               // row tile in (B*T)
    const int n0 = blockIdx.y * BN;               // col tile in E

    // row stride 40 halves = 80B -> every 8-half chunk is 16B aligned
    __shared__ _Float16 As[BM][BK + 8];
    __shared__ _Float16 Bs[BN][BK + 8];

    const int tid   = threadIdx.x;                // 0..127 (4 waves)
    const int lane  = tid & 31;
    const int wave  = tid >> 5;
    const int lhalf = lane & 15;
    const int lhi   = lane >> 4;

    v8f acc[4] = {};                              // 16(M) x 64(N) per wave

    for (int k0 = 0; k0 < kE; k0 += BK) {
        __syncthreads();
        // Stage X and W tiles (64x32 f32 each) into LDS as f16.
        for (int i = 0; i < 4; ++i) {
            int idx = tid + 128 * i;              // 0..511
            int r   = idx >> 3;                   // row 0..63
            int c4  = idx & 7;                    // float4 within row
            v4f x = *(const v4f*)(X + (size_t)(m0 + r) * kE + k0 + c4 * 4);
            As[r][c4 * 4 + 0] = (_Float16)x.x;
            As[r][c4 * 4 + 1] = (_Float16)x.y;
            As[r][c4 * 4 + 2] = (_Float16)x.z;
            As[r][c4 * 4 + 3] = (_Float16)x.w;
            v4f w = *(const v4f*)(W + (size_t)(n0 + r) * kE + k0 + c4 * 4);
            Bs[r][c4 * 4 + 0] = (_Float16)w.x;
            Bs[r][c4 * 4 + 1] = (_Float16)w.y;
            Bs[r][c4 * 4 + 2] = (_Float16)w.z;
            Bs[r][c4 * 4 + 3] = (_Float16)w.w;
        }
        __syncthreads();

        // A fragment for this wave's 16 rows.
        v16h a;
        {
            const _Float16* row = &As[16 * wave + lhalf][0];
            v8h c0 = *(const v8h*)(row + 16 * lhi);       // halves 0..7
            v8h c1 = *(const v8h*)(row + 16 * lhi + 8);   // halves 8..15
            for (int i = 0; i < 8; ++i) { a[i] = c0[i]; a[8 + i] = c1[i]; }
        }
        // 4 N-tiles, A fragment reused.
        for (int nt = 0; nt < 4; ++nt) {
            const _Float16* brow = &Bs[nt * 16 + lhalf][0];
            v8h b0 = *(const v8h*)(brow + 16 * lhi);
            v8h b1 = *(const v8h*)(brow + 16 * lhi + 8);
            v16h bfrag;
            for (int i = 0; i < 8; ++i) { bfrag[i] = b0[i]; bfrag[8 + i] = b1[i]; }
            acc[nt] = __builtin_amdgcn_wmma_f32_16x16x32_f16(
                false, a, false, bfrag, (short)0, acc[nt], false, false);
        }
    }

    // Store D tile as f16 (row = m0+16w+g+8*lhi, col = n0+16nt+lhalf).
    for (int nt = 0; nt < 4; ++nt)
        for (int g = 0; g < 8; ++g) {
            int row = m0 + 16 * wave + g + 8 * lhi;
            int col = n0 + nt * 16 + lhalf;
            Dst[(size_t)row * kE + col] = (_Float16)acc[nt][g];
        }
}

// ---- Kernel 2: causal flash attention -------------------------------------
// Block = 128 threads = 4 waves; blockIdx.x = b*H + h; blockIdx.y = q-group.
// Wave w owns q-tile qt = 4*qgrp + w. All waves share one async-staged V chunk
// per 32-wide KV block; each wave has a private P staging slab.
__global__ __launch_bounds__(128)
void attn_kernel(const _Float16* __restrict__ Qh, const _Float16* __restrict__ Kh,
                 const _Float16* __restrict__ Vh, float* __restrict__ Out)
{
    const int bh    = blockIdx.x;                 // b*H + h
    const int b     = bh >> 4;
    const int h     = bh & 15;
    const int qgrp  = blockIdx.y;                 // 0..31 (4 q-tiles each)
    const int tid   = threadIdx.x;
    const int lane  = tid & 31;
    const int wave  = tid >> 5;
    const int lhalf = lane & 15;
    const int lhi   = lane >> 4;
    const int qt    = qgrp * 4 + wave;            // this wave's q-tile

    __shared__ _Float16 Vs[32][kD + 8];           // shared V chunk, row 144B
    __shared__ _Float16 Ps[4][16][32 + 8];        // per-wave P tile, row 80B

    // Q A-fragments (K = d 0..31 and 32..63) kept in registers for whole sweep.
    const _Float16* qrow =
        Qh + ((size_t)(b * kT + qt * 16 + lhalf) * kE + h * kD);
    v16h qa[2];
    for (int c = 0; c < 2; ++c) {
        v8h c0 = *(const v8h*)(qrow + 32 * c + 16 * lhi);
        v8h c1 = *(const v8h*)(qrow + 32 * c + 16 * lhi + 8);
        for (int i = 0; i < 8; ++i) { qa[c][i] = c0[i]; qa[c][8 + i] = c1[i]; }
    }

    v8f  o[4] = {};                               // O accumulator 16 x 64
    float mrow[8], lrow[8];
    for (int g = 0; g < 8; ++g) { mrow[g] = -1e30f; lrow[g] = 0.f; }

    const int my_jmax  = (qt * 16 + 15) >> 5;     // last KV block for this wave
    const int grp_jmax = 2 * qgrp + 1;            // last KV block for the group

    for (int j = 0; j <= grp_jmax; ++j) {
        const int kv0 = j * 32;
        const bool active = (j <= my_jmax);       // wave-uniform guard
        __syncthreads();                          // prev Vs readers done

        // Async-stage the shared V chunk (32 kv rows x 64 d f16 = 4 KB):
        // 128 threads x 2 x 16B, ASYNCcnt-tracked, overlaps the S WMMAs below.
        {
            const int r = tid >> 2, c = tid & 3;  // row 0..31, 16B chunk 0..3
            const _Float16* vsrc =
                Vh + ((size_t)(b * kT + kv0 + r) * kE + h * kD) + 8 * c;
            unsigned lds0 = (unsigned)(uintptr_t)&Vs[r][8 * c];
            asm volatile("global_load_async_to_lds_b128 %0, %1, off"
                         :: "v"(lds0), "v"((unsigned long long)(uintptr_t)vsrc)
                         : "memory");
            asm volatile("global_load_async_to_lds_b128 %0, %1, off"
                         :: "v"(lds0 + 64u),
                            "v"((unsigned long long)(uintptr_t)(vsrc + 32))
                         : "memory");
        }

        if (active) {
            // S = Q @ K^T for two 16-wide kv sub-tiles (B-frag straight from L2).
            v8f s[2] = {};
            for (int st = 0; st < 2; ++st) {
                const _Float16* krow =
                    Kh + ((size_t)(b * kT + kv0 + st * 16 + lhalf) * kE + h * kD);
                for (int c = 0; c < 2; ++c) {
                    v8h b0 = *(const v8h*)(krow + 32 * c + 16 * lhi);
                    v8h b1 = *(const v8h*)(krow + 32 * c + 16 * lhi + 8);
                    v16h kb;
                    for (int i = 0; i < 8; ++i) { kb[i] = b0[i]; kb[8 + i] = b1[i]; }
                    s[st] = __builtin_amdgcn_wmma_f32_16x16x32_f16(
                        false, qa[c], false, kb, (short)0, s[st], false, false);
                }
            }

            // Streaming softmax over the 16x32 block.
            const float scale = 0.125f;           // 1/sqrt(64)
            for (int g = 0; g < 8; ++g) {
                const int q  = qt * 16 + g + 8 * lhi;
                float s0 = s[0][g] * scale;
                float s1 = s[1][g] * scale;
                if (kv0 + lhalf      > q) s0 = -1e30f;
                if (kv0 + 16 + lhalf > q) s1 = -1e30f;
                float mx = fmaxf(s0, s1);
                for (int off = 1; off < 16; off <<= 1)    // reduce 16-lane half
                    mx = fmaxf(mx, __shfl_xor(mx, off, 32));
                const float mnew  = fmaxf(mrow[g], mx);
                const float alpha = __expf(mrow[g] - mnew);
                const float p0 = __expf(s0 - mnew);
                const float p1 = __expf(s1 - mnew);
                float ps = p0 + p1;
                for (int off = 1; off < 16; off <<= 1)
                    ps += __shfl_xor(ps, off, 32);
                lrow[g] = lrow[g] * alpha + ps;
                mrow[g] = mnew;
                for (int dt = 0; dt < 4; ++dt) o[dt][g] *= alpha;
                Ps[wave][g + 8 * lhi][lhalf]      = (_Float16)p0;
                Ps[wave][g + 8 * lhi][16 + lhalf] = (_Float16)p1;
            }
        }

        // Drain our async V stores, then publish to all waves.
        asm volatile("s_wait_asynccnt 0x0" ::: "memory");
        __syncthreads();

        if (active) {
            // P A-fragment (16x32) from this wave's LDS slab.
            v16h pa;
            {
                const _Float16* prow = &Ps[wave][lhalf][0];
                v8h c0 = *(const v8h*)(prow + 16 * lhi);
                v8h c1 = *(const v8h*)(prow + 16 * lhi + 8);
                for (int i = 0; i < 8; ++i) { pa[i] = c0[i]; pa[8 + i] = c1[i]; }
            }
            // O += P @ V (4 d-tiles) from the shared V chunk.
            for (int dt = 0; dt < 4; ++dt) {
                v16h vb;                          // B 32x16: N=d fixed per lane
                for (int i = 0; i < 16; ++i)
                    vb[i] = Vs[16 * lhi + i][dt * 16 + lhalf];
                o[dt] = __builtin_amdgcn_wmma_f32_16x16x32_f16(
                    false, pa, false, vb, (short)0, o[dt], false, false);
            }
        }
    }

    // Epilogue: O / l, f32 out.
    for (int g = 0; g < 8; ++g) {
        const float inv = 1.0f / lrow[g];
        const size_t row = (size_t)(b * kT + qt * 16 + g + 8 * lhi);
        for (int dt = 0; dt < 4; ++dt)
            Out[row * kE + h * kD + dt * 16 + lhalf] = o[dt][g] * inv;
    }
}

// ---------------------------------------------------------------------------
extern "C" void kernel_launch(void* const* d_in, const int* in_sizes, int n_in,
                              void* d_out, int out_size, void* d_ws, size_t ws_size,
                              hipStream_t stream) {
    const float* Xq  = (const float*)d_in[0];
    const float* Xkv = (const float*)d_in[1];
    const float* Wq  = (const float*)d_in[2];
    const float* Wk  = (const float*)d_in[3];
    const float* Wv  = (const float*)d_in[4];
    float* Out = (float*)d_out;

    // Workspace: f16 Q/K/V, 16 MB each (48 MB total).
    const size_t proj_elems = (size_t)kB * kT * kE;
    _Float16* Qh = (_Float16*)d_ws;
    _Float16* Kh = Qh + proj_elems;
    _Float16* Vh = Kh + proj_elems;

    dim3 g1(kB * kT / BM, kE / BN, 3);            // 128 x 16 x 3
    proj_kernel<<<g1, 128, 0, stream>>>(Xq, Xkv, Wq, Wk, Wv, Qh, Kh, Vh);

    dim3 g2(kB * kH, kT / 16 / 4);                // 64 x 32 blocks, 4 waves each
    attn_kernel<<<g2, 128, 0, stream>>>(Qh, Kh, Vh, Out);
}
